// WorldGraphEncoder_46694884442137
// MI455X (gfx1250) — matrix operand order repaired and checked by reading, not compiled, not run on hardware
//
#include <hip/hip_runtime.h>
#include <hip/hip_bf16.h>
#include <stdint.h>

typedef unsigned short u16;
typedef __attribute__((ext_vector_type(8)))  __bf16 bf16x8;
typedef __attribute__((ext_vector_type(16))) __bf16 bf16x16;
typedef __attribute__((ext_vector_type(8)))  float  f32x8;

static __device__ __forceinline__ u16 f2bf(float f) {
  unsigned u = __float_as_uint(f);
  unsigned r = u + 0x7fffu + ((u >> 16) & 1u);   // round-to-nearest-even
  return (u16)(r >> 16);
}
static __device__ __forceinline__ unsigned fkey(float f) {
  unsigned u = __float_as_uint(f);
  return (u & 0x80000000u) ? ~u : (u | 0x80000000u);   // monotone float->uint
}
static __device__ __forceinline__ float funkey(unsigned k) {
  return __uint_as_float((k & 0x80000000u) ? (k & 0x7fffffffu) : ~k);
}
static __device__ __forceinline__ float lrelu(float x) { return x > 0.f ? x : 0.2f * x; }

// ---------------- utility kernels ----------------

__global__ void fill_u32_k(unsigned* p, unsigned v, size_t n) {
  size_t i = (size_t)blockIdx.x * blockDim.x + threadIdx.x;
  size_t stride = (size_t)gridDim.x * blockDim.x;
  for (; i < n; i += stride) p[i] = v;
}

__global__ void to_bf16_k(const float* __restrict__ x, u16* __restrict__ y, size_t n) {
  size_t i = (size_t)blockIdx.x * blockDim.x + threadIdx.x;
  size_t stride = (size_t)gridDim.x * blockDim.x;
  for (; i < n; i += stride) y[i] = f2bf(x[i]);
}

// W [K,Nc] row-major f32  ->  Wt [Nc,K] row-major bf16
__global__ void transpose_bf16_k(const float* __restrict__ W, u16* __restrict__ Wt,
                                 int K, int Nc) {
  int i = blockIdx.x * blockDim.x + threadIdx.x;
  if (i >= K * Nc) return;
  int k = i / Nc, n = i % Nc;
  Wt[(size_t)n * K + k] = f2bf(W[i]);
}

// ---------------- bf16 WMMA GEMM: C[M,Nc] = A[M,K] @ W[K,Nc] (W stored transposed) ----
// block = 256 threads (8 waves); wave w -> 16x16 tile at (blockIdx.x*16, blockIdx.y*128+w*16)
// M % 16 == 0, K % 32 == 0, Nc % 128 == 0 (per grid.y), all guaranteed by caller.
__global__ __launch_bounds__(256) void gemm_bf16_wmma_k(
    const u16* __restrict__ A, const u16* __restrict__ Bt,
    const float* __restrict__ bias, float* __restrict__ outf,
    u16* __restrict__ outbf, int M, int K, int Nc, int relu_act)
{
  const int lane = threadIdx.x & 31;
  const int wave = threadIdx.x >> 5;
  const int m0 = blockIdx.x * 16;
  const int n0 = blockIdx.y * 128 + wave * 16;
  const int half = lane >> 4;        // 0: lanes 0-15, 1: lanes 16-31
  const int l15  = lane & 15;
  const int arow = m0 + l15;         // A fragment row
  const int bcol = n0 + l15;         // B fragment col

  f32x8 c = {0.f, 0.f, 0.f, 0.f, 0.f, 0.f, 0.f, 0.f};

  for (int k0 = 0; k0 < K; k0 += 32) {
    // A 16x32 bf16 fragment: elems 0-7 -> K = k0+half*8.. ; elems 8-15 -> K = k0+16+half*8..
    const u16* ap = A + (size_t)arow * K + k0 + half * 8;
    bf16x8 a_lo = *reinterpret_cast<const bf16x8*>(ap);
    bf16x8 a_hi = *reinterpret_cast<const bf16x8*>(ap + 16);
    bf16x16 a = __builtin_shufflevector(a_lo, a_hi,
        0, 1, 2, 3, 4, 5, 6, 7, 8, 9, 10, 11, 12, 13, 14, 15);
    // B 32x16 bf16 fragment: lane holds col bcol, elems 0-7 -> K = k0+half*8.. ; elems 8-15 -> K = k0+16+half*8..
    const u16* bp = Bt + (size_t)bcol * K + k0 + half * 8;
    bf16x8 b_lo = *reinterpret_cast<const bf16x8*>(bp);
    bf16x8 b_hi = *reinterpret_cast<const bf16x8*>(bp + 16);
    bf16x16 b = __builtin_shufflevector(b_lo, b_hi,
        0, 1, 2, 3, 4, 5, 6, 7, 8, 9, 10, 11, 12, 13, 14, 15);
    c = __builtin_amdgcn_wmma_f32_16x16x32_bf16(
        /*neg_a=*/false, a, /*neg_b=*/false, b,
        /*c_mod=*/(short)0, c, /*reuse_a=*/false, /*reuse_b=*/false);
  }

#pragma unroll
  for (int r = 0; r < 8; ++r) {
    int row = m0 + r + half * 8;     // C layout: VGPR r -> M = r + 8*(lane>>4)
    int col = bcol;
    float v = c[r];
    if (bias) v += bias[col];
    if (relu_act) v = v > 0.f ? v : 0.f;
    size_t idx = (size_t)row * Nc + col;
    if (outf)  outf[idx]  = v;
    if (outbf) outbf[idx] = f2bf(v);
  }
}

// ---------------- attention scores: a[n,h] = sum_c h[n,h,c] * att[h,c] ----------------
__global__ void attn_reduce_k(const float* __restrict__ h, const float* __restrict__ att,
                              float* __restrict__ outv, int N, int C) {
  int i = blockIdx.x * blockDim.x + threadIdx.x;   // over N*4
  if (i >= N * 4) return;
  int hh = i & 3;
  const float* hp = h + (size_t)i * C;
  const float* ap = att + hh * C;
  float s = 0.f;
  for (int c = 0; c < C; ++c) s += hp[c] * ap[c];
  outv[i] = s;
}

// ---------------- edge kernels (edge list + N appended self-loops) ----------------
static __device__ __forceinline__ void edge_sd(const int* ei, int E, int i, int& s, int& d) {
  if (i < E) { s = ei[i]; d = ei[E + i]; } else { s = d = i - E; }
}

__global__ void edge_max_k(const int* __restrict__ ei, int E, int N,
                           const float* __restrict__ as, const float* __restrict__ ad,
                           unsigned* __restrict__ m) {
  int i = blockIdx.x * blockDim.x + threadIdx.x;
  if (i >= E + N) return;
  int s, d; edge_sd(ei, E, i, s, d);
#pragma unroll
  for (int h = 0; h < 4; ++h) {
    float e = lrelu(as[s * 4 + h] + ad[d * 4 + h]);
    atomicMax(&m[d * 4 + h], fkey(e));
  }
}

__global__ void edge_sum_k(const int* __restrict__ ei, int E, int N,
                           const float* __restrict__ as, const float* __restrict__ ad,
                           const unsigned* __restrict__ m, float* __restrict__ ssum) {
  int i = blockIdx.x * blockDim.x + threadIdx.x;
  if (i >= E + N) return;
  int s, d; edge_sd(ei, E, i, s, d);
#pragma unroll
  for (int h = 0; h < 4; ++h) {
    float e = lrelu(as[s * 4 + h] + ad[d * 4 + h]);
    atomicAdd(&ssum[d * 4 + h], expf(e - funkey(m[d * 4 + h])));
  }
}

// one thread per (edge, head); alpha folded in -> single scatter pass
__global__ void edge_scatter_k(const int* __restrict__ ei, int E, int N,
                               const float* __restrict__ as, const float* __restrict__ ad,
                               const unsigned* __restrict__ m, const float* __restrict__ ssum,
                               const float* __restrict__ hsrc, int C, int concat,
                               float* __restrict__ outp) {
  int t = blockIdx.x * blockDim.x + threadIdx.x;
  int tot = (E + N) * 4;
  if (t >= tot) return;
  int i = t >> 2, h = t & 3;
  int s, d; edge_sd(ei, E, i, s, d);
  float e = lrelu(as[s * 4 + h] + ad[d * 4 + h]);
  float alpha = expf(e - funkey(m[d * 4 + h])) / (ssum[d * 4 + h] + 1e-16f);
  const float* hp = hsrc + ((size_t)s * 4 + h) * C;
  float* op = outp + (size_t)d * (concat ? 4 * C : C) + (concat ? (size_t)h * C : 0);
  for (int c = 0; c < C; ++c) atomicAdd(op + c, hp[c] * alpha);
}

// ---------------- bias + BatchNorm(eval) + ELU -> bf16 for next GEMM ----------------
__global__ void norm_elu_k(const float* __restrict__ agg, const float* __restrict__ bias,
                           const float* __restrict__ g, const float* __restrict__ b,
                           u16* __restrict__ xbf, int N) {
  int i = blockIdx.x * blockDim.x + threadIdx.x;
  if (i >= N * 128) return;
  int c = i & 127;
  float v = agg[i] + bias[c];
  v = v * 0.99999500003749979f * g[c] + b[c];   // 1/sqrt(1+1e-5)
  v = v > 0.f ? v : expm1f(v);
  xbf[i] = f2bf(v);
}

// ---------------- column mean (partial per block, atomic combine) ----------------
__global__ void colmean_k(const float* __restrict__ agg, float* __restrict__ acc, int N) {
  int j = threadIdx.x;                 // 128 threads = 128 columns
  int r0 = blockIdx.x * 100;
  float s = 0.f;
  for (int r = 0; r < 100; ++r) {
    int row = r0 + r;
    if (row < N) s += agg[(size_t)row * 128 + j];
  }
  atomicAdd(&acc[j], s);
}

// ---------------- final head: relu(mean @ out_W + out_b) ----------------
__global__ void final_k(const float* __restrict__ acc, const float* __restrict__ b2,
                        const float* __restrict__ Wo, const float* __restrict__ bo,
                        float* __restrict__ out, int N) {
  __shared__ float mf[128];
  int j = threadIdx.x;
  mf[j] = acc[j] * (0.25f / (float)N) + b2[j];   // /4 heads, /N nodes, + bias2
  __syncthreads();
  float y = bo[j];
  for (int c = 0; c < 128; ++c) y += mf[c] * Wo[c * 128 + j];
  out[j] = y > 0.f ? y : 0.f;
}

// ---------------- host launcher ----------------
extern "C" void kernel_launch(void* const* d_in, const int* in_sizes, int n_in,
                              void* d_out, int out_size, void* d_ws, size_t ws_size,
                              hipStream_t stream) {
  const float* emb   = (const float*)d_in[0];
  const int*   ei    = (const int*)  d_in[1];
  const float* projW = (const float*)d_in[2];
  const float* projb = (const float*)d_in[3];
  const float* W1    = (const float*)d_in[4];
  const float* att_s1= (const float*)d_in[5];
  const float* att_d1= (const float*)d_in[6];
  const float* b1    = (const float*)d_in[7];
  const float* gam   = (const float*)d_in[8];
  const float* bet   = (const float*)d_in[9];
  const float* W2    = (const float*)d_in[10];
  const float* att_s2= (const float*)d_in[11];
  const float* att_d2= (const float*)d_in[12];
  const float* b2    = (const float*)d_in[13];
  const float* outW  = (const float*)d_in[14];
  const float* outb  = (const float*)d_in[15];
  float* out = (float*)d_out;

  const int N  = in_sizes[0] / 128;
  const int E  = in_sizes[1] / 2;
  const int ET = E + N;

  char* ws = (char*)d_ws;
  size_t cur = 0;
  auto alloc = [&](size_t bytes) -> char* {
    char* p = ws + cur;
    cur = (cur + bytes + 255) & ~(size_t)255;
    return p;
  };

  // persistent across whole pipeline
  u16*   wt0    = (u16*)  alloc((size_t)128 * 128 * 2);
  u16*   wt1    = (u16*)  alloc((size_t)128 * 128 * 2);
  u16*   wt2    = (u16*)  alloc((size_t)512 * 128 * 2);
  u16*   x2bf   = (u16*)  alloc((size_t)N * 128 * 2);
  float* agg2   = (float*)alloc((size_t)N * 128 * 4);
  float* macc   = (float*)alloc(128 * 4);
  size_t arena = cur;
  // arena layout A (layer 1)
  u16*   embbf  = (u16*)  alloc((size_t)N * 128 * 2);
  u16*   xbf    = (u16*)  alloc((size_t)N * 128 * 2);
  float* h1     = (float*)alloc((size_t)N * 128 * 4);
  float* a1s    = (float*)alloc((size_t)N * 4 * 4);
  float* a1d    = (float*)alloc((size_t)N * 4 * 4);
  unsigned* m1  = (unsigned*)alloc((size_t)N * 4 * 4);
  float* s1     = (float*)alloc((size_t)N * 4 * 4);
  float* agg1   = (float*)alloc((size_t)N * 128 * 4);
  // arena layout B (layer 2) — overlays layout A, which is dead by then
  cur = arena;
  float* h2     = (float*)alloc((size_t)N * 512 * 4);
  float* a2s    = (float*)alloc((size_t)N * 4 * 4);
  float* a2d    = (float*)alloc((size_t)N * 4 * 4);
  unsigned* m2  = (unsigned*)alloc((size_t)N * 4 * 4);
  float* s2     = (float*)alloc((size_t)N * 4 * 4);

  const unsigned KEY_NEG_INF = 0x007FFFFFu;   // fkey(-inf)
  dim3 blk(256);

  // weight prep (f32 -> transposed bf16)
  transpose_bf16_k<<<(128 * 128 + 255) / 256, blk, 0, stream>>>(projW, wt0, 128, 128);
  transpose_bf16_k<<<(128 * 128 + 255) / 256, blk, 0, stream>>>(W1,    wt1, 128, 128);
  transpose_bf16_k<<<(128 * 512 + 255) / 256, blk, 0, stream>>>(W2,    wt2, 128, 512);
  to_bf16_k<<<2048, blk, 0, stream>>>(emb, embbf, (size_t)N * 128);

  // layer-1 accumulator init
  fill_u32_k<<<1024, blk, 0, stream>>>((unsigned*)s1,  0u,          (size_t)N * 4);
  fill_u32_k<<<1024, blk, 0, stream>>>(m1,             KEY_NEG_INF, (size_t)N * 4);
  fill_u32_k<<<2048, blk, 0, stream>>>((unsigned*)agg1, 0u,         (size_t)N * 128);

  // x = relu(emb @ proj_W + proj_b)  (bf16 out only)
  dim3 g1(N / 16, 1);
  gemm_bf16_wmma_k<<<g1, blk, 0, stream>>>(embbf, wt0, projb, nullptr, xbf, N, 128, 128, 1);
  // h1 = x @ W1
  gemm_bf16_wmma_k<<<g1, blk, 0, stream>>>(xbf, wt1, nullptr, h1, nullptr, N, 128, 128, 0);

  // attention scores layer 1
  attn_reduce_k<<<(N * 4 + 255) / 256, blk, 0, stream>>>(h1, att_s1, a1s, N, 32);
  attn_reduce_k<<<(N * 4 + 255) / 256, blk, 0, stream>>>(h1, att_d1, a1d, N, 32);

  // segment softmax + aggregate (layer 1, concat heads)
  int eb = (ET + 255) / 256;
  edge_max_k<<<eb, blk, 0, stream>>>(ei, E, N, a1s, a1d, m1);
  edge_sum_k<<<eb, blk, 0, stream>>>(ei, E, N, a1s, a1d, m1, s1);
  edge_scatter_k<<<(ET * 4 + 255) / 256, blk, 0, stream>>>(ei, E, N, a1s, a1d, m1, s1,
                                                           h1, 32, 1, agg1);
  // +bias1, BN(eval), ELU -> bf16
  norm_elu_k<<<(N * 128 + 255) / 256, blk, 0, stream>>>(agg1, b1, gam, bet, x2bf, N);

  // layer-2 accumulator init (arena A now dead)
  fill_u32_k<<<1024, blk, 0, stream>>>((unsigned*)s2,   0u,          (size_t)N * 4);
  fill_u32_k<<<1024, blk, 0, stream>>>(m2,              KEY_NEG_INF, (size_t)N * 4);
  fill_u32_k<<<2048, blk, 0, stream>>>((unsigned*)agg2, 0u,          (size_t)N * 128);
  fill_u32_k<<<1, 128, 0, stream>>>((unsigned*)macc,    0u,          128);

  // h2 = x2 @ W2  [N,512]
  dim3 g2(N / 16, 4);
  gemm_bf16_wmma_k<<<g2, blk, 0, stream>>>(x2bf, wt2, nullptr, h2, nullptr, N, 128, 512, 0);

  // attention scores layer 2
  attn_reduce_k<<<(N * 4 + 255) / 256, blk, 0, stream>>>(h2, att_s2, a2s, N, 128);
  attn_reduce_k<<<(N * 4 + 255) / 256, blk, 0, stream>>>(h2, att_d2, a2d, N, 128);

  // segment softmax + aggregate (layer 2, sum over heads into [N,128]; /4 deferred)
  edge_max_k<<<eb, blk, 0, stream>>>(ei, E, N, a2s, a2d, m2);
  edge_sum_k<<<eb, blk, 0, stream>>>(ei, E, N, a2s, a2d, m2, s2);
  edge_scatter_k<<<(ET * 4 + 255) / 256, blk, 0, stream>>>(ei, E, N, a2s, a2d, m2, s2,
                                                           h2, 128, 0, agg2);

  // node mean + final dense head
  colmean_k<<<(N + 99) / 100, 128, 0, stream>>>(agg2, macc, N);
  final_k<<<1, 128, 0, stream>>>(macc, b2, outW, outb, out, N);
}
